// DynamicNet_40089224741417
// MI455X (gfx1250) — matrix operand order repaired and checked by reading, compiled, not run on hardware
//
#include <hip/hip_runtime.h>

typedef __attribute__((ext_vector_type(16))) __bf16 v16bf;
typedef __attribute__((ext_vector_type(8)))  float  v8f;
typedef int gv4i __attribute__((vector_size(16)));   // matches builtin param type

#define Bn 512
#define Tn 128
#define Dn 300
#define Dp 320
#define Hn 512
#define Mn 1024
#define Cn 3

// ---------------- async bf16 tile copy: global -> LDS (ASYNCcnt path) ----------
__device__ __forceinline__ void async_copy_b128(const void* g, void* l) {
#if defined(__AMDGCN__) && __has_builtin(__builtin_amdgcn_global_load_async_to_lds_b128)
  __builtin_amdgcn_global_load_async_to_lds_b128(
      (__attribute__((address_space(1))) gv4i*)(gv4i*)(void*)g,
      (__attribute__((address_space(3))) gv4i*)(gv4i*)l,
      /*imm offset=*/0, /*cpol=*/0);
#else
  *(uint4*)l = *(const uint4*)g;
#endif
}

__device__ __forceinline__ void wait_async0() {
#if defined(__AMDGCN__) && __has_builtin(__builtin_amdgcn_global_load_async_to_lds_b128)
#if __has_builtin(__builtin_amdgcn_s_wait_asynccnt)
  __builtin_amdgcn_s_wait_asynccnt(0);
#else
  asm volatile("s_wait_asynccnt 0x0" ::: "memory");
#endif
#endif
}

// ---------------- fragment loaders (ISA 7.12.2 layouts, wave32) ----------------
// A 16-bit 16x32: lanes 0-15 -> K{0..7,16..23}, lanes 16-31 -> K{8..15,24..31}
__device__ __forceinline__ v16bf load_frag_a(const __bf16* rowptr, int half) {
  const uint32_t* p = (const uint32_t*)rowptr;
  union { uint32_t u[8]; v16bf v; } t;
  int b0 = half * 4;        // dwords: K0..7 or K8..15
  int b1 = 8 + half * 4;    // dwords: K16..23 or K24..31
#pragma unroll
  for (int i = 0; i < 4; ++i) { t.u[i] = p[b0 + i]; t.u[4 + i] = p[b1 + i]; }
  return t.v;
}
// B 16-bit 32x16: lanes 0-15 -> K{0..7,16..23}, lanes 16-31 -> K{8..15,24..31}
__device__ __forceinline__ v16bf load_frag_b(const __bf16* rowptr, int half) {
  const uint32_t* p = (const uint32_t*)rowptr;
  union { uint32_t u[8]; v16bf v; } t;
#pragma unroll
  for (int i = 0; i < 4; ++i) { t.u[i] = p[half * 4 + i]; t.u[4 + i] = p[8 + half * 4 + i]; }
  return t.v;
}

// ---------------- prep: f32 -> bf16 weight conversion + h0 = 0 ----------------
__global__ __launch_bounds__(256) void prep_kernel(
    const float* __restrict__ W_ih, const float* __restrict__ W_hh,
    const float* __restrict__ l0_w,
    __bf16* __restrict__ Wih_bf, __bf16* __restrict__ Whh_bf,
    __bf16* __restrict__ l0w_bf, float* __restrict__ hA) {
  int i = blockIdx.x * blockDim.x + threadIdx.x;
  if (i < Hn * Dp) {
    int r = i / Dp, d = i % Dp;
    Wih_bf[i] = (d < Dn) ? (__bf16)W_ih[r * Dn + d] : (__bf16)0.0f;
  }
  if (i < Hn * Hn) { Whh_bf[i] = (__bf16)W_hh[i]; hA[i] = 0.0f; }
  if (i < Mn * Hn) { l0w_bf[i] = (__bf16)l0_w[i]; }
}

// ---------------- one RNN step: h_dst = mask ? tanh(E[tok]Wih^T + h Whh^T + b) : h_src
// grid (8,8): x = 64-col tile of H, y = 64-row tile of B. 256 threads = 8 waves,
// each wave computes a 16x32 strip via two v_wmma_f32_16x16x32_bf16 accumulators.
__global__ __launch_bounds__(256) void rnn_step_kernel(
    const int* __restrict__ x, const int* __restrict__ lengths,
    const float* __restrict__ E,
    const __bf16* __restrict__ Wih_bf, const __bf16* __restrict__ Whh_bf,
    const float* __restrict__ b_ih, const float* __restrict__ b_hh,
    const float* __restrict__ h_src, float* __restrict__ h_dst, int t) {
  __shared__ alignas(16) __bf16 Asb[64][40];  // [batch row][k], +8 pad
  __shared__ alignas(16) __bf16 Bsb[64][40];  // [out col n][k], +8 pad

  const int tid  = threadIdx.x;
  const int lane = tid & 31, wave = tid >> 5;
  const int half = lane >> 4, rr = lane & 15;
  const int wrow = (wave & 3) * 16;   // 0..48
  const int wcol = (wave >> 2) * 32;  // 0 or 32
  const int n0 = blockIdx.x * 64;
  const int b0 = blockIdx.y * 64;

  v8f acc0 = {0.f,0.f,0.f,0.f,0.f,0.f,0.f,0.f};
  v8f acc1 = {0.f,0.f,0.f,0.f,0.f,0.f,0.f,0.f};

  const int lrow = tid >> 2;        // 0..63 (row of A tile / col of B tile)
  const int koff = (tid & 3) * 8;   // 0,8,16,24

  const int tok = x[(b0 + lrow) * Tn + (Tn - 1 - t)];
  const float*  erow = E + (long)tok * Dn;
  const float*  hrow = h_src + (b0 + lrow) * Hn;
  const __bf16* wi   = Wih_bf + (n0 + lrow) * Dp;
  const __bf16* wh   = Whh_bf + (n0 + lrow) * Hn;

  __builtin_prefetch(erow + koff, 0, 0);  // global_prefetch_b8: warm gather lines

  for (int kb = 0; kb < 26; ++kb) {
    if (kb < 10) {                       // embedding-projection K range (padded 320)
      int kg = kb * 32 + koff;
      async_copy_b128(wi + kg, &Bsb[lrow][koff]);   // weights: pure bf16 copy
#pragma unroll
      for (int j = 0; j < 8; ++j) {
        int k = kg + j;
        float f = (k < Dn) ? erow[k] : 0.0f;
        Asb[lrow][koff + j] = (__bf16)f;
      }
    } else {                             // recurrent K range
      int kg = (kb - 10) * 32 + koff;
      async_copy_b128(wh + kg, &Bsb[lrow][koff]);
#pragma unroll
      for (int j = 0; j < 8; ++j) Asb[lrow][koff + j] = (__bf16)hrow[kg + j];
    }
    wait_async0();       // our async LDS writes done before publishing
    __syncthreads();
    v16bf af  = load_frag_a(&Asb[wrow + rr][0], half);
    v16bf bf0 = load_frag_b(&Bsb[wcol + rr][0], half);
    v16bf bf1 = load_frag_b(&Bsb[wcol + 16 + rr][0], half);
    acc0 = __builtin_amdgcn_wmma_f32_16x16x32_bf16(false, af, false, bf0,
                                                   (short)0, acc0, false, false);
    acc1 = __builtin_amdgcn_wmma_f32_16x16x32_bf16(false, af, false, bf1,
                                                   (short)0, acc1, false, false);
    __syncthreads();
  }

  const int nc0 = n0 + wcol + rr;
  const int nc1 = nc0 + 16;
  const float bias0 = b_ih[nc0] + b_hh[nc0];
  const float bias1 = b_ih[nc1] + b_hh[nc1];
  const int rbase = b0 + wrow + half * 8;   // C/D: row = vgpr + 8*half, col = lane%16
#pragma unroll
  for (int g = 0; g < 8; ++g) {
    int row = rbase + g;
    bool act = (Tn - 1 - t) < lengths[row];
    float v0 = tanhf(acc0[g] + bias0);
    float v1 = tanhf(acc1[g] + bias1);
    h_dst[row * Hn + nc0] = act ? v0 : h_src[row * Hn + nc0];
    h_dst[row * Hn + nc1] = act ? v1 : h_src[row * Hn + nc1];
  }
}

// ---------------- MLP layer 0: act = relu(h @ l0_w^T + l0_b), WMMA GEMM K=512
__global__ __launch_bounds__(256) void mlp1_kernel(
    const float* __restrict__ h, const __bf16* __restrict__ l0w_bf,
    const float* __restrict__ l0_b, float* __restrict__ act) {
  __shared__ alignas(16) __bf16 Asb[64][40];
  __shared__ alignas(16) __bf16 Bsb[64][40];

  const int tid  = threadIdx.x;
  const int lane = tid & 31, wave = tid >> 5;
  const int half = lane >> 4, rr = lane & 15;
  const int wrow = (wave & 3) * 16;
  const int wcol = (wave >> 2) * 32;
  const int n0 = blockIdx.x * 64;   // 0..1023
  const int b0 = blockIdx.y * 64;   // 0..511

  v8f acc0 = {0.f,0.f,0.f,0.f,0.f,0.f,0.f,0.f};
  v8f acc1 = {0.f,0.f,0.f,0.f,0.f,0.f,0.f,0.f};

  const int lrow = tid >> 2;
  const int koff = (tid & 3) * 8;
  const float*  hrow = h + (b0 + lrow) * Hn;
  const __bf16* wr   = l0w_bf + (n0 + lrow) * Hn;

  for (int kb = 0; kb < 16; ++kb) {
    int kg = kb * 32 + koff;
    async_copy_b128(wr + kg, &Bsb[lrow][koff]);
#pragma unroll
    for (int j = 0; j < 8; ++j) Asb[lrow][koff + j] = (__bf16)hrow[kg + j];
    wait_async0();
    __syncthreads();
    v16bf af  = load_frag_a(&Asb[wrow + rr][0], half);
    v16bf bf0 = load_frag_b(&Bsb[wcol + rr][0], half);
    v16bf bf1 = load_frag_b(&Bsb[wcol + 16 + rr][0], half);
    acc0 = __builtin_amdgcn_wmma_f32_16x16x32_bf16(false, af, false, bf0,
                                                   (short)0, acc0, false, false);
    acc1 = __builtin_amdgcn_wmma_f32_16x16x32_bf16(false, af, false, bf1,
                                                   (short)0, acc1, false, false);
    __syncthreads();
  }

  const int nc0 = n0 + wcol + rr;
  const int nc1 = nc0 + 16;
  const float bias0 = l0_b[nc0], bias1 = l0_b[nc1];
  const int rbase = b0 + wrow + half * 8;
#pragma unroll
  for (int g = 0; g < 8; ++g) {
    int row = rbase + g;
    act[row * Mn + nc0] = fmaxf(acc0[g] + bias0, 0.0f);
    act[row * Mn + nc1] = fmaxf(acc1[g] + bias1, 0.0f);
  }
}

// ---------------- MLP layer 1 (C=3) + log_softmax: one wave per batch row
__global__ __launch_bounds__(128) void mlp2_kernel(
    const float* __restrict__ act, const float* __restrict__ l1_w,
    const float* __restrict__ l1_b, float* __restrict__ out) {
  const int wave = threadIdx.x >> 5, lane = threadIdx.x & 31;
  const int b = blockIdx.x * 4 + wave;
  const float* ar = act + b * Mn;
  float s0 = 0.f, s1 = 0.f, s2 = 0.f;
  for (int m = lane; m < Mn; m += 32) {
    float a = ar[m];
    s0 += a * l1_w[m];
    s1 += a * l1_w[Mn + m];
    s2 += a * l1_w[2 * Mn + m];
  }
#pragma unroll
  for (int off = 16; off > 0; off >>= 1) {
    s0 += __shfl_xor(s0, off, 32);
    s1 += __shfl_xor(s1, off, 32);
    s2 += __shfl_xor(s2, off, 32);
  }
  if (lane == 0) {
    float l0v = fmaxf(s0 + l1_b[0], 0.0f);
    float l1v = fmaxf(s1 + l1_b[1], 0.0f);
    float l2v = fmaxf(s2 + l1_b[2], 0.0f);
    float mx  = fmaxf(l0v, fmaxf(l1v, l2v));
    float lse = logf(expf(l0v - mx) + expf(l1v - mx) + expf(l2v - mx));
    out[b * Cn + 0] = l0v - mx - lse;
    out[b * Cn + 1] = l1v - mx - lse;
    out[b * Cn + 2] = l2v - mx - lse;
  }
}

// ---------------- launcher ----------------
extern "C" void kernel_launch(void* const* d_in, const int* in_sizes, int n_in,
                              void* d_out, int out_size, void* d_ws, size_t ws_size,
                              hipStream_t stream) {
  const int*   x       = (const int*)d_in[0];
  const int*   lengths = (const int*)d_in[1];
  const float* E       = (const float*)d_in[2];
  const float* W_ih    = (const float*)d_in[3];
  const float* b_ih    = (const float*)d_in[4];
  const float* W_hh    = (const float*)d_in[5];
  const float* b_hh    = (const float*)d_in[6];
  const float* l0_w    = (const float*)d_in[7];
  const float* l0_b    = (const float*)d_in[8];
  const float* l1_w    = (const float*)d_in[9];
  const float* l1_b    = (const float*)d_in[10];
  (void)in_sizes; (void)n_in; (void)out_size; (void)ws_size;

  char* ws = (char*)d_ws;                 // regions, all 256B-aligned offsets
  __bf16* Wih_bf = (__bf16*)(ws + 0);             // 512*320*2  = 327680
  __bf16* Whh_bf = (__bf16*)(ws + 327680);        // 512*512*2  = 524288
  __bf16* l0w_bf = (__bf16*)(ws + 851968);        // 1024*512*2 = 1048576
  float*  hA     = (float*) (ws + 1900544);       // 512*512*4  = 1048576
  float*  hB     = (float*) (ws + 2949120);       // 512*512*4  = 1048576
  float*  act    = (float*) (ws + 3997696);       // 512*1024*4 = 2097152
  float*  out    = (float*)d_out;

  prep_kernel<<<2048, 256, 0, stream>>>(W_ih, W_hh, l0_w, Wih_bf, Whh_bf, l0w_bf, hA);

  for (int t = 0; t < Tn; ++t) {
    const float* src = (t & 1) ? hB : hA;
    float*       dst = (t & 1) ? hA : hB;
    rnn_step_kernel<<<dim3(8, 8), 256, 0, stream>>>(
        x, lengths, E, Wih_bf, Whh_bf, b_ih, b_hh, src, dst, t);
  }
  // 128 steps: final write lands in hA

  mlp1_kernel<<<dim3(16, 8), 256, 0, stream>>>(hA, l0w_bf, l0_b, act);
  mlp2_kernel<<<128, 128, 0, stream>>>(act, l1_w, l1_b, out);
}
